// FeatureNetLoss_64561948394251
// MI455X (gfx1250) — matrix operand order repaired and checked by reading, not compiled, not run on hardware
//
#include <hip/hip_runtime.h>
#include <hip/hip_bf16.h>
#include <math.h>

// ---------------- problem constants (from reference setup_inputs) -----------
#define BB    6
#define NPTS  1024
#define DDIM  128
#define HH    320
#define WW    320
#define HW    (HH*WW)
#define NC    500
#define MTOT  (BB*NPTS)      // 6144
#define NTIL  (MTOT/16)      // 384
#define BROW  136            // padded LDS row stride (halves) to spread banks

typedef _Float16 v16h __attribute__((ext_vector_type(16)));
typedef _Float16 v8h  __attribute__((ext_vector_type(8)));
typedef float    v8f  __attribute__((ext_vector_type(8)));

// ---------------- small device helpers --------------------------------------
__device__ __forceinline__ void inv3(const float* K, float* Ki) {
    float a=K[0],b=K[1],c=K[2],d=K[3],e=K[4],f=K[5],g=K[6],h=K[7],i=K[8];
    float A =  e*i - f*h;
    float Bc = -(d*i - f*g);
    float C =  d*h - e*g;
    float det = a*A + b*Bc + c*C;
    float id = 1.0f/det;
    Ki[0]=A*id;            Ki[1]=-(b*i-c*h)*id;  Ki[2]=(b*f-c*e)*id;
    Ki[3]=Bc*id;           Ki[4]=(a*i-c*g)*id;   Ki[5]=-(a*f-c*d)*id;
    Ki[6]=C*id;            Ki[7]=-(a*h-b*g)*id;  Ki[8]=(a*e-b*d)*id;
}

__device__ __forceinline__ unsigned okey(float f) {
    unsigned u = __float_as_uint(f);
    return (u & 0x80000000u) ? ~u : (u | 0x80000000u); // order-preserving map
}

// A/B fragment gather for v_wmma_f32_16x16x32_f16 (wave32 layout):
// lanes 0-15 (hi=0): halves 0..7 = K[k0..k0+7],    halves 8..15 = K[k0+16..k0+23]
// lanes16-31 (hi=1): halves 0..7 = K[k0+8..k0+15], halves 8..15 = K[k0+24..k0+31]
__device__ __forceinline__ v16h load_frag(const _Float16* row, int k0, int hi) {
    v8h lo = *(const v8h*)(row + k0 + hi*8);
    v8h hh = *(const v8h*)(row + k0 + 16 + hi*8);
    v16h r;
#pragma unroll
    for (int i = 0; i < 8; ++i) { r[i] = lo[i]; r[i+8] = hh[i]; }
    return r;
}

// ---------------- K1: normalize descriptors -> f16 --------------------------
__global__ void k_norm(const float* __restrict__ desc, _Float16* __restrict__ X) {
    int tid = blockIdx.x*blockDim.x + threadIdx.x;
    if (tid >= MTOT) return;
    const float* v = desc + (size_t)tid*DDIM;
    float ss = 0.f;
#pragma unroll 8
    for (int i = 0; i < DDIM; ++i) { float x = v[i]; ss += x*x; }
    float inv = rsqrtf(fmaxf(ss, 1e-16f));
    _Float16* o = X + (size_t)tid*DDIM;
#pragma unroll 8
    for (int i = 0; i < DDIM; ++i) o[i] = (_Float16)(v[i]*inv);
}

// ---------------- K2: pairwise_project of the N input points ----------------
// srcpx[b,n,2], dstpx[a,b,n,2] = (u,v) in pixels, vis[a,b,n]
__global__ void k_proj(const float* __restrict__ points, const float* __restrict__ depths,
                       const float* __restrict__ poses,  const float* __restrict__ Km,
                       float* __restrict__ srcpx, float* __restrict__ dstpx,
                       int* __restrict__ visb) {
    int tid = blockIdx.x*blockDim.x + threadIdx.x;
    if (tid >= BB*NPTS) return;
    int b = tid / NPTS, n = tid % NPTS;
    float Ki[9]; inv3(Km, Ki);
    float px = (points[tid*2+0] + 1.f)*0.5f*(WW-1);
    float py = (points[tid*2+1] + 1.f)*0.5f*(HH-1);
    srcpx[tid*2+0] = px; srcpx[tid*2+1] = py;
    int xi = min(max((int)rintf(px), 0), WW-1);
    int yi = min(max((int)rintf(py), 0), HH-1);
    float d = depths[b*HW + yi*WW + xi];
    float cx = (Ki[0]*px + Ki[1]*py + Ki[2]) * d;
    float cy = (Ki[3]*px + Ki[4]*py + Ki[5]) * d;
    float cz = (Ki[6]*px + Ki[7]*py + Ki[8]) * d;
    const float* Pb = poses + b*16;
    float wx = Pb[0]*cx + Pb[1]*cy + Pb[2]*cz  + Pb[3];
    float wy = Pb[4]*cx + Pb[5]*cy + Pb[6]*cz  + Pb[7];
    float wz = Pb[8]*cx + Pb[9]*cy + Pb[10]*cz + Pb[11];
    for (int a = 0; a < BB; ++a) {
        const float* Pa = poses + a*16;
        float dx = wx - Pa[3], dy = wy - Pa[7], dz = wz - Pa[11];
        float c0 = Pa[0]*dx + Pa[4]*dy + Pa[8]*dz;   // R^T * diff
        float c1 = Pa[1]*dx + Pa[5]*dy + Pa[9]*dz;
        float c2 = Pa[2]*dx + Pa[6]*dy + Pa[10]*dz;
        float u0 = Km[0]*c0 + Km[1]*c1 + Km[2]*c2;
        float v0 = Km[3]*c0 + Km[4]*c1 + Km[5]*c2;
        float z  = Km[6]*c0 + Km[7]*c1 + Km[8]*c2;
        float zs = fabsf(z) > 1e-6f ? z : 1e-6f;
        float u = u0/zs, v = v0/zs;
        int vis = (z > 0.1f) && (u >= 0.f) && (u <= (float)(WW-1))
                             && (v >= 0.f) && (v <= (float)(HH-1));
        int o = (a*BB + b)*NPTS + n;
        dstpx[o*2+0] = u; dstpx[o*2+1] = v;
        visb[o] = vis;
    }
}

// ---------------- K3: WMMA Gram GEMM with fused masked reductions -----------
// Block = 8 waves. blockIdx.x selects a 4-tile B panel (64 rows of X) shared by
// all 8 waves and staged once in LDS; blockIdx.y*8+wave selects the A tile.
// Each wave computes a 16x64 strip: A fragment loaded once per k-step and
// reused across 4 WMMAs (register blocking).  ~44 FLOP/byte from L2.
__global__ __launch_bounds__(256) void k_gemm(const _Float16* __restrict__ X,
        const float* __restrict__ srcpx, const float* __restrict__ dstpx,
        const int* __restrict__ visb, float* __restrict__ acc,
        unsigned int* __restrict__ cnt) {
    __shared__ _Float16 Bsh[64*BROW];      // 17,408 B of the 320 KB WGP pool

    int lane = threadIdx.x & 31;
    int wave = threadIdx.x >> 5;
    int tile_m = blockIdx.y * 8 + wave;    // 0..383
    int tn0    = blockIdx.x * 4;           // first of 4 column tiles
    int hi  = lane >> 4;
    int r16 = lane & 15;

    // cooperative stage of the 64-row B panel (1024 x 16B chunks, 4 per thread)
    int rowbase_n = tn0 * 16;
    for (int c = threadIdx.x; c < 1024; c += 256) {
        int r = c >> 4, s = c & 15;
        *(v8h*)&Bsh[r*BROW + s*8] =
            *(const v8h*)(X + (size_t)(rowbase_n + r)*DDIM + s*8);
    }
    __syncthreads();

    const _Float16* arow = X + (size_t)(tile_m*16 + r16)*DDIM;

    v8f c4[4] = {};
#pragma unroll
    for (int k0 = 0; k0 < DDIM; k0 += 32) {
        v16h a = load_frag(arow, k0, hi);
#pragma unroll
        for (int t = 0; t < 4; ++t) {
            v16h b = load_frag(&Bsh[(t*16 + r16)*BROW], k0, hi);
            c4[t] = __builtin_amdgcn_wmma_f32_16x16x32_f16(
                        false, a, false, b, (short)0, c4[t], false, false);
        }
    }

    // epilogue: D layout -> row = vgpr + 8*hi, col = lane&15
    int ab = tile_m >> 6;                  // row batch a (N=1024 -> 64 tiles)
    int bb = tn0 >> 6;                     // col batch b (4-tile group never straddles)
    int base_m = (tile_m & 63)*16;
    bool diag = (ab == bb);
    const int pairbase = (ab*BB + bb)*NPTS;

    int   vism[8]; float sxm[8], sym[8];
#pragma unroll
    for (int v = 0; v < 8; ++v) {
        int mloc = base_m + v + 8*hi;
        vism[v] = visb[pairbase + mloc];
        sxm[v]  = srcpx[(bb*NPTS + mloc)*2+0];
        sym[v]  = srcpx[(bb*NPTS + mloc)*2+1];
    }

    float dsum = 0.f, psum = 0.f, nsum = 0.f;
    int pc = 0, ncnt = 0;
#pragma unroll
    for (int t = 0; t < 4; ++t) {
        int nloc = ((tn0 + t) & 63)*16 + r16;
        float du = dstpx[(pairbase + nloc)*2+0];
        float dv = dstpx[(pairbase + nloc)*2+1];
#pragma unroll
        for (int v = 0; v < 8; ++v) {
            int mloc = base_m + v + 8*hi;
            float s = c4[t][v];
            if (diag) dsum += fmaxf(s, 0.f);
            if (mloc < nloc && vism[v]) {
                float ex = sxm[v] - du, ey = sym[v] - dv;
                float dist = sqrtf(ex*ex + ey*ey);
                if (dist <= 1.0f) { psum += s; ++pc; }
                else              { nsum += s; ++ncnt; }
            }
        }
    }
#pragma unroll
    for (int off = 16; off; off >>= 1) {
        dsum += __shfl_down(dsum, off, 32);
        psum += __shfl_down(psum, off, 32);
        nsum += __shfl_down(nsum, off, 32);
        pc   += __shfl_down(pc,   off, 32);
        ncnt += __shfl_down(ncnt, off, 32);
    }
    if (lane == 0) {
        if (diag) atomicAdd(&acc[0], dsum);
        atomicAdd(&acc[1], psum);
        atomicAdd(&acc[2], nsum);
        atomicAdd(&cnt[0], (unsigned)pc);
        atomicAdd(&cnt[1], (unsigned)ncnt);
    }
}

// ---------------- image pipeline --------------------------------------------
__global__ void k_gray(const float* __restrict__ imgs, float* __restrict__ gray) {
    int tid = blockIdx.x*blockDim.x + threadIdx.x;
    if (tid >= BB*HW) return;
    int b = tid / HW, p = tid % HW;
    const float* ib = imgs + (size_t)b*3*HW;
    gray[tid] = 0.299f*ib[p] + 0.587f*ib[HW+p] + 0.114f*ib[2*HW+p];
}

__global__ void k_sobel(const float* __restrict__ gray,
                        float* __restrict__ gx, float* __restrict__ gy) {
    int tid = blockIdx.x*blockDim.x + threadIdx.x;
    if (tid >= BB*HW) return;
    int b = tid / HW, p = tid % HW, y = p / WW, x = p % WW;
    const float sx[3][3] = {{-0.125f,0.f,0.125f},{-0.25f,0.f,0.25f},{-0.125f,0.f,0.125f}};
    const float* g = gray + (size_t)b*HW;
    float ax = 0.f, ay = 0.f;
#pragma unroll
    for (int i = 0; i < 3; ++i) {
        int ry = min(max(y + i - 1, 0), HH-1);
#pragma unroll
        for (int j = 0; j < 3; ++j) {
            int rx = min(max(x + j - 1, 0), WW-1);
            float v = g[ry*WW + rx];
            ax += sx[i][j]*v;
            ay += sx[j][i]*v;   // transpose
        }
    }
    gx[tid] = ax; gy[tid] = ay;
}

// 7x7 gaussian-smoothed structure tensor -> min-eigenvalue response (reflect)
__global__ void k_resp(const float* __restrict__ gx, const float* __restrict__ gy,
                       float* __restrict__ resp) {
    int tid = blockIdx.x*blockDim.x + threadIdx.x;
    if (tid >= BB*HW) return;
    int b = tid / HW, p = tid % HW, y = p / WW, x = p % WW;
    float gs[7], gsum = 0.f;
#pragma unroll
    for (int i = 0; i < 7; ++i) { float a = (float)(i-3); gs[i] = expf(-0.5f*a*a); gsum += gs[i]; }
    float inv2 = 1.f/(gsum*gsum);
    const float* px = gx + (size_t)b*HW;
    const float* py = gy + (size_t)b*HW;
    float sxx = 0.f, syy = 0.f, sxy = 0.f;
    for (int i = 0; i < 7; ++i) {
        int ry = y + i - 3; if (ry < 0) ry = -ry; if (ry >= HH) ry = 2*HH - 2 - ry;
        for (int j = 0; j < 7; ++j) {
            int rx = x + j - 3; if (rx < 0) rx = -rx; if (rx >= WW) rx = 2*WW - 2 - rx;
            float w = gs[i]*gs[j]*inv2;
            float a = px[ry*WW + rx], c = py[ry*WW + rx];
            sxx += w*a*a; syy += w*c*c; sxy += w*a*c;
        }
    }
    float tr = sxx + syy;
    float dd = sxx - syy;
    float disc = sqrtf(fmaxf(dd*dd + 4.f*sxy*sxy, 0.f));
    resp[tid] = 0.5f*(tr - disc);
}

__global__ void k_nms5(const float* __restrict__ in, float* __restrict__ out) {
    int tid = blockIdx.x*blockDim.x + threadIdx.x;
    if (tid >= BB*HW) return;
    int b = tid / HW, p = tid % HW, y = p / WW, x = p % WW;
    const float* s = in + (size_t)b*HW;
    float m = -INFINITY;
    for (int i = -2; i <= 2; ++i) {
        int ry = y + i; if (ry < 0 || ry >= HH) continue;
        for (int j = -2; j <= 2; ++j) {
            int rx = x + j; if (rx < 0 || rx >= WW) continue;
            m = fmaxf(m, s[ry*WW + rx]);
        }
    }
    float v = s[p];
    out[tid] = (v == m) ? v : 0.f;
}

__global__ void k_blockkeep(const float* __restrict__ in, float* __restrict__ out) {
    int tid = blockIdx.x*blockDim.x + threadIdx.x;
    if (tid >= BB*HW) return;
    int b = tid / HW, p = tid % HW, y = p / WW, x = p % WW;
    const float* s = in + (size_t)b*HW;
    int by = y & ~7, bx = x & ~7;
    float m = -INFINITY;
    for (int i = 0; i < 8; ++i)
        for (int j = 0; j < 8; ++j)
            m = fmaxf(m, s[(by+i)*WW + (bx+j)]);
    float v = s[p];
    out[tid] = (v == m) ? v : 0.f;
}

// ---------------- per-batch top-500 via 1024-bin radix select ---------------
__global__ __launch_bounds__(1024) void k_topk(const float* __restrict__ keep,
                                               int* __restrict__ cidx,
                                               float* __restrict__ cval) {
    __shared__ unsigned hist[1024];
    __shared__ unsigned sT, sNgt, cgt, ctie;
    int b = blockIdx.x, t = threadIdx.x;
    hist[t] = 0;
    if (t == 0) { cgt = 0; ctie = 0; }
    __syncthreads();
    const float* kb = keep + (size_t)b*HW;
    for (int i = t; i < HW; i += 1024)
        atomicAdd(&hist[okey(kb[i]) >> 22], 1u);
    __syncthreads();
    if (t == 0) {
        unsigned cum = 0, T = 0, ngt = 0;
        for (int k = 1023; k >= 0; --k) {
            if (cum + hist[k] >= NC) { T = (unsigned)k; ngt = cum; break; }
            cum += hist[k];
        }
        sT = T; sNgt = ngt;
    }
    __syncthreads();
    unsigned T = sT, ngt = sNgt;
    for (int i = t; i < HW; i += 1024) {
        float v = kb[i];
        unsigned k = okey(v) >> 22;
        if (k > T) {
            unsigned s = atomicAdd(&cgt, 1u);
            cidx[b*NC + s] = i; cval[b*NC + s] = v;
        } else if (k == T) {
            unsigned s = atomicAdd(&ctie, 1u);
            if (ngt + s < NC) { cidx[b*NC + ngt + s] = i; cval[b*NC + ngt + s] = v; }
        }
    }
}

// ---------------- project corners, scatter-max into target ------------------
// thread (a,b,n): corner n of batch b projected into view a; value = cval[a,n]
__global__ void k_scatter(const int* __restrict__ cidx, const float* __restrict__ cval,
                          const float* __restrict__ depths, const float* __restrict__ poses,
                          const float* __restrict__ Km, float* __restrict__ target) {
    int tid = blockIdx.x*blockDim.x + threadIdx.x;
    if (tid >= BB*BB*NC) return;
    int n  = tid % NC;
    int ab = tid / NC;
    int b  = ab % BB;
    int a  = ab / BB;
    float val = cval[a*NC + n];
    if (val <= 0.f) return;
    int idx = cidx[b*NC + n];
    int ix = idx % WW, iy = idx / WW;
    float Ki[9]; inv3(Km, Ki);
    float px = (float)ix, py = (float)iy;
    float d = depths[b*HW + iy*WW + ix];
    float cx = (Ki[0]*px + Ki[1]*py + Ki[2]) * d;
    float cy = (Ki[3]*px + Ki[4]*py + Ki[5]) * d;
    float cz = (Ki[6]*px + Ki[7]*py + Ki[8]) * d;
    const float* Pb = poses + b*16;
    float wx = Pb[0]*cx + Pb[1]*cy + Pb[2]*cz  + Pb[3];
    float wy = Pb[4]*cx + Pb[5]*cy + Pb[6]*cz  + Pb[7];
    float wz = Pb[8]*cx + Pb[9]*cy + Pb[10]*cz + Pb[11];
    const float* Pa = poses + a*16;
    float dx = wx - Pa[3], dy = wy - Pa[7], dz = wz - Pa[11];
    float c0 = Pa[0]*dx + Pa[4]*dy + Pa[8]*dz;
    float c1 = Pa[1]*dx + Pa[5]*dy + Pa[9]*dz;
    float c2 = Pa[2]*dx + Pa[6]*dy + Pa[10]*dz;
    float u0 = Km[0]*c0 + Km[1]*c1 + Km[2]*c2;
    float v0 = Km[3]*c0 + Km[4]*c1 + Km[5]*c2;
    float z  = Km[6]*c0 + Km[7]*c1 + Km[8]*c2;
    float zs = fabsf(z) > 1e-6f ? z : 1e-6f;
    float u = u0/zs, v = v0/zs;
    bool vis = (z > 0.1f) && (u >= 0.f) && (u <= (float)(WW-1))
                          && (v >= 0.f) && (v <= (float)(HH-1));
    if (!vis) return;
    int wf = (int)rintf(u), hf = (int)rintf(v);
    if (wf < 0 || hf < 0 || wf >= WW || hf >= HH) return;
    atomicMax((unsigned*)&target[b*HW + hf*WW + wf], __float_as_uint(val)); // val > 0
}

// ---------------- BCE + laplacian regularizer (fused with target NMS) -------
__global__ void k_bce(const float* __restrict__ target, const float* __restrict__ scores,
                      float* __restrict__ acc) {
    int tid = blockIdx.x*blockDim.x + threadIdx.x;
    int b = tid / HW, p = tid % HW, y = p / WW, x = p % WW;
    const float* tg = target + (size_t)b*HW;
    const float* sc = scores + (size_t)b*HW;
    // nms2d(target,5) then >0
    float tv = tg[p];
    float m = -INFINITY;
    for (int i = -2; i <= 2; ++i) {
        int ry = y + i; if (ry < 0 || ry >= HH) continue;
        for (int j = -2; j <= 2; ++j) {
            int rx = x + j; if (rx < 0 || rx >= WW) continue;
            m = fmaxf(m, tg[ry*WW + rx]);
        }
    }
    float tb = (tv == m && tv > 0.f) ? 1.f : 0.f;
    float pr = fminf(fmaxf(sc[p], 1e-12f), 1.f - 1e-7f);
    float bce = -(tb*logf(pr) + (1.f - tb)*logf(1.f - pr));
    // laplacian5 (reflect): center -0.5, others 1/48
    float lap = 0.f;
    for (int i = -2; i <= 2; ++i) {
        int ry = y + i; if (ry < 0) ry = -ry; if (ry >= HH) ry = 2*HH - 2 - ry;
        for (int j = -2; j <= 2; ++j) {
            int rx = x + j; if (rx < 0) rx = -rx; if (rx >= WW) rx = 2*WW - 2 - rx;
            float w = (i == 0 && j == 0) ? -0.5f : (1.f/48.f);
            lap += w*sc[ry*WW + rx];
        }
    }
    float reg = sc[p]*expf(-lap);
#pragma unroll
    for (int off = 16; off; off >>= 1) {
        bce += __shfl_down(bce, off, 32);
        reg += __shfl_down(reg, off, 32);
    }
    if ((threadIdx.x & 31) == 0) {
        atomicAdd(&acc[3], bce);
        atomicAdd(&acc[4], reg);
    }
}

// ---------------- finalize ---------------------------------------------------
__global__ void k_final(const float* __restrict__ acc, const unsigned* __restrict__ cnt,
                        float* __restrict__ out) {
    if (blockIdx.x == 0 && threadIdx.x == 0) {
        float distinction = acc[0] / (float)(BB*NPTS*NPTS);
        float posm = acc[1] / (float)(cnt[0] > 0u ? cnt[0] : 1u);
        float negm = acc[2] / (float)(cnt[1] > 0u ? cnt[1] : 1u);
        float match = 1.f - posm + negm;
        float bce = acc[3] / (float)(BB*HW);
        float reg = acc[4] / (float)(BB*HW);
        float cornerness = bce + reg*10.f;
        out[0] = 1.0f*distinction + 0.5f*cornerness + 1.0f*match;
    }
}

// ---------------- host launcher ---------------------------------------------
extern "C" void kernel_launch(void* const* d_in, const int* in_sizes, int n_in,
                              void* d_out, int out_size, void* d_ws, size_t ws_size,
                              hipStream_t stream) {
    const float* desc   = (const float*)d_in[0];   // (6,1024,128)
    const float* points = (const float*)d_in[1];   // (6,1024,2)
    const float* scores = (const float*)d_in[2];   // (6,1,320,320)
    const float* depths = (const float*)d_in[3];   // (6,1,320,320)
    const float* poses  = (const float*)d_in[4];   // (6,4,4)
    const float* Km     = (const float*)d_in[5];   // (3,3)
    const float* imgs   = (const float*)d_in[6];   // (6,3,320,320)

    char* ws = (char*)d_ws;
    const size_t IMGB = (size_t)BB*HW*sizeof(float);   // 2,457,600 B
    size_t o = 0;
    _Float16* X     = (_Float16*)(ws + o); o += (size_t)MTOT*DDIM*2;   // 1.5 MB
    float*    srcpx = (float*)(ws + o);    o += (size_t)BB*NPTS*2*4;
    float*    dstpx = (float*)(ws + o);    o += (size_t)BB*BB*NPTS*2*4;
    int*      visb  = (int*)(ws + o);      o += (size_t)BB*BB*NPTS*4;
    int*      cidx  = (int*)(ws + o);      o += 12288;
    float*    cval  = (float*)(ws + o);    o += 12288;
    float*    acc   = (float*)(ws + o);
    unsigned* cnt   = (unsigned*)(ws + o + 32); o += 256;
    float* gray = (float*)(ws + o); o += IMGB;   // reused as corners
    float* gxb  = (float*)(ws + o); o += IMGB;   // reused as keep
    float* gyb  = (float*)(ws + o); o += IMGB;
    float* resp = (float*)(ws + o); o += IMGB;   // reused as target
    float* corners = gray;
    float* keep    = gxb;
    float* target  = resp;

    hipMemsetAsync(acc, 0, 256, stream);

    // descriptor Gram path
    k_norm<<<MTOT/256, 256, 0, stream>>>(desc, X);
    k_proj<<<(BB*NPTS)/256, 256, 0, stream>>>(points, depths, poses, Km,
                                              srcpx, dstpx, visb);
    k_gemm<<<dim3(NTIL/4, NTIL/8), 256, 0, stream>>>(X, srcpx, dstpx, visb, acc, cnt);

    // image path
    const int PG = (BB*HW)/256;   // 2400, exact
    k_gray<<<PG, 256, 0, stream>>>(imgs, gray);
    k_sobel<<<PG, 256, 0, stream>>>(gray, gxb, gyb);
    k_resp<<<PG, 256, 0, stream>>>(gxb, gyb, resp);
    k_nms5<<<PG, 256, 0, stream>>>(resp, corners);
    k_blockkeep<<<PG, 256, 0, stream>>>(corners, keep);
    k_topk<<<BB, 1024, 0, stream>>>(keep, cidx, cval);
    hipMemsetAsync(target, 0, IMGB, stream);
    k_scatter<<<(BB*BB*NC + 255)/256, 256, 0, stream>>>(cidx, cval, depths, poses,
                                                        Km, target);
    k_bce<<<PG, 256, 0, stream>>>(target, scores, acc);

    k_final<<<1, 32, 0, stream>>>(acc, cnt, (float*)d_out);
}
